// RNN_50087908606352
// MI455X (gfx1250) — matrix operand order repaired and checked by reading, not compiled, not run on hardware
//
#include <hip/hip_runtime.h>

#define Lx 1024
#define Bn 128
#define In 256
#define Hn 512
#define NWG 16

typedef __attribute__((ext_vector_type(16))) __bf16 v16bf;
typedef __attribute__((ext_vector_type(8)))  float  v8f;
typedef __attribute__((ext_vector_type(4)))  int    v4i;

union Frag {
  unsigned int u[8];
  unsigned short s[16];
  v16bf v;
};

__device__ __forceinline__ unsigned short f2bf(float f) {
  unsigned int u = __float_as_uint(f);
  u += 0x7FFFu + ((u >> 16) & 1u);   // round-to-nearest-even
  return (unsigned short)(u >> 16);
}
__device__ __forceinline__ unsigned int pack2(float a, float b) {
  return (unsigned int)f2bf(a) | ((unsigned int)f2bf(b) << 16);
}

// ---- async global->LDS staging (CDNA5 GLOBAL_LOAD_ASYNC_TO_LDS_B128) ------
#if __has_builtin(__builtin_amdgcn_global_load_async_to_lds_b128) && \
    __has_builtin(__builtin_amdgcn_s_wait_asynccnt)
#define USE_ASYNC_LDS 1
#else
#define USE_ASYNC_LDS 0
#endif

typedef __attribute__((address_space(1))) v4i* gv4i_p;  // global src
typedef __attribute__((address_space(3))) v4i* lv4i_p;  // LDS dst

__device__ __forceinline__ void copy16_g2l(const void* g, void* l) {
#if USE_ASYNC_LDS
  // flat->AS1 is numerically identity; flat LDS addr[31:0] == LDS offset (ISA 10.2)
  __builtin_amdgcn_global_load_async_to_lds_b128(
      (gv4i_p)(uintptr_t)g, (lv4i_p)(unsigned int)(uintptr_t)l, 0, 0);
#else
  *(uint4*)l = *(const uint4*)g;
#endif
}
__device__ __forceinline__ void async_join() {
#if USE_ASYNC_LDS
  __builtin_amdgcn_s_wait_asynccnt(0);
#endif
}

// ---------------- prep: fp32 -> bf16 for Wx, Wh, h0; zero sync counter ------
__global__ void prep_kernel(const float* __restrict__ h0,
                            const float* __restrict__ Wx,
                            const float* __restrict__ Wh,
                            unsigned short* __restrict__ wx_bf,
                            unsigned short* __restrict__ wh_bf,
                            unsigned short* __restrict__ h_bf0,
                            int* __restrict__ syncc) {
  int idx = blockIdx.x * blockDim.x + threadIdx.x;
  if (idx == 0) syncc[0] = 0;
  if (idx < Hn * In) wx_bf[idx] = f2bf(Wx[idx]);
  if (idx < Hn * Hn) wh_bf[idx] = f2bf(Wh[idx]);
  if (idx < Bn * Hn) h_bf0[idx] = f2bf(h0[idx]);
}

// ---------------- phase 1: xproj = x @ Wx^T  -> d_out (fp32) ----------------
// WG tile: M=32 rows of (L*B), N=512 (all of H), K looped 256 in steps of 32.
__global__ __launch_bounds__(256) void xproj_kernel(
    const float* __restrict__ x, const unsigned short* __restrict__ wx_bf,
    float* __restrict__ out) {
  __shared__ unsigned short As[32][40];    // 32x32 bf16 tile of x (padded)
  __shared__ unsigned short Bs[512][40];   // 512x32 bf16 tile of Wx (padded)
  const int tid = threadIdx.x;
  const int wave = tid >> 5, lane = tid & 31;
  const int lm = lane & 15, hi = lane >> 4;
  const int kbA = hi * 8, kbB = hi * 16;
  const int m0 = blockIdx.x * 32;

  v8f zero = {};
  v8f acc[2][4];
  for (int i = 0; i < 2; ++i)
    for (int j = 0; j < 4; ++j) acc[i][j] = zero;

  for (int k0 = 0; k0 < In; k0 += 32) {
    // stage B: all 512 rows of Wx_bf, 32 k (8 x 16B per thread, async)
    for (int i = 0; i < 8; ++i) {
      int idx = tid + 256 * i;
      int row = idx >> 2, seg = idx & 3;
      copy16_g2l(&wx_bf[(size_t)row * In + k0 + seg * 8], &Bs[row][seg * 8]);
    }
    // stage A: 32 rows x 32 k, fp32->bf16 on the fly (1 float4 per thread)
    {
      int row = tid >> 3, seg = tid & 7;
      const float4 f =
          *(const float4*)&x[(size_t)(m0 + row) * In + k0 + seg * 4];
      *(uint2*)&As[row][seg * 4] = make_uint2(pack2(f.x, f.y), pack2(f.z, f.w));
    }
    async_join();
    __syncthreads();

    Frag a[2];
    for (int mb = 0; mb < 2; ++mb) {
      const unsigned short* ap = &As[mb * 16 + lm][0];
      for (int p = 0; p < 4; ++p) {
        a[mb].u[p]     = *(const unsigned int*)&ap[kbA + 2 * p];
        a[mb].u[4 + p] = *(const unsigned int*)&ap[16 + kbA + 2 * p];
      }
    }
    for (int nb = 0; nb < 4; ++nb) {
      Frag b;
      const unsigned short* bp = &Bs[wave * 64 + nb * 16 + lm][0];
      for (int p = 0; p < 4; ++p) {
        b.u[p]     = *(const unsigned int*)&bp[kbA + 2 * p];
        b.u[4 + p] = *(const unsigned int*)&bp[16 + kbA + 2 * p];
      }
      for (int mb = 0; mb < 2; ++mb)
        acc[mb][nb] = __builtin_amdgcn_wmma_f32_16x16x32_bf16(
            false, a[mb].v, false, b.v, (short)0, acc[mb][nb], false, false);
    }
    __syncthreads();
  }

  for (int mb = 0; mb < 2; ++mb)
    for (int nb = 0; nb < 4; ++nb) {
      int col = wave * 64 + nb * 16 + lm;
      for (int r = 0; r < 8; ++r) {
        int row = m0 + mb * 16 + r + hi * 8;
        out[(size_t)row * Hn + col] = acc[mb][nb][r];
      }
    }
}

// ---------------- phase 2: persistent recurrence ----------------------------
// 16 WGs, WG b owns H columns [32b, 32b+32). Wh slice (32x512 bf16) and the
// ENTIRE h state (128x512 bf16) live in dynamic LDS (166 KB of the 320 KB/WGP)
// -> exactly one stage + one barrier per serial timestep.
#define P2PAD 520
#define P2SMEM ((32 + 128) * P2PAD * 2)

__global__ __launch_bounds__(256) void rnn_kernel(
    const float* __restrict__ bh, const unsigned short* __restrict__ wh_bf,
    unsigned short* __restrict__ hbuf0, unsigned short* __restrict__ hbuf1,
    float* __restrict__ out, int* __restrict__ syncc) {
  extern __shared__ unsigned short smem[];
  unsigned short* WhS  = smem;                 // [32][P2PAD]
  unsigned short* hAll = smem + 32 * P2PAD;    // [128][P2PAD]

  const int tid = threadIdx.x;
  const int wave = tid >> 5, lane = tid & 31;
  const int lm = lane & 15, hi = lane >> 4;
  const int kbA = hi * 8, kbB = hi * 16;
  const int c0 = blockIdx.x * 32;

  // preload Wh slice once: 32x512 bf16 (8 x 16B per thread, async)
  for (int i = 0; i < 8; ++i) {
    int idx = tid + 256 * i;
    int row = idx >> 6, seg = idx & 63;
    copy16_g2l(&wh_bf[(size_t)(c0 + row) * Hn + seg * 8],
               &WhS[row * P2PAD + seg * 8]);
  }
  float bias[2];
  for (int nb = 0; nb < 2; ++nb) bias[nb] = bh[c0 + nb * 16 + lm];
  async_join();
  __syncthreads();

  for (int t = 0; t < Lx; ++t) {
    const unsigned short* hsrc = (t & 1) ? hbuf1 : hbuf0;
    unsigned short* hdst       = (t & 1) ? hbuf0 : hbuf1;

    // stage full h state: 128x512 bf16 (32 x 16B per thread, async)
    for (int i = 0; i < 32; ++i) {
      int idx = tid + 256 * i;
      int row = idx >> 6, seg = idx & 63;
      copy16_g2l(&hsrc[(size_t)row * Hn + seg * 8],
                 &hAll[row * P2PAD + seg * 8]);
    }
    async_join();
    __syncthreads();

    v8f zero = {};
    v8f acc[2];
    acc[0] = zero; acc[1] = zero;

#pragma unroll 4
    for (int k0 = 0; k0 < Hn; k0 += 32) {
      Frag a;
      const unsigned short* ap = &hAll[(wave * 16 + lm) * P2PAD + k0];
      for (int p = 0; p < 4; ++p) {
        a.u[p]     = *(const unsigned int*)&ap[kbA + 2 * p];
        a.u[4 + p] = *(const unsigned int*)&ap[16 + kbA + 2 * p];
      }
      for (int nb = 0; nb < 2; ++nb) {
        Frag b;
        const unsigned short* bp = &WhS[(nb * 16 + lm) * P2PAD + k0];
        for (int p = 0; p < 4; ++p) {
          b.u[p]     = *(const unsigned int*)&bp[kbA + 2 * p];
          b.u[4 + p] = *(const unsigned int*)&bp[16 + kbA + 2 * p];
        }
        acc[nb] = __builtin_amdgcn_wmma_f32_16x16x32_bf16(
            false, a.v, false, b.v, (short)0, acc[nb], false, false);
      }
    }

    // prefetch next timestep's xproj slice (one 128B line covers our 32 cols)
    if (t + 1 < Lx && tid < Bn)
      __builtin_prefetch(&out[((size_t)(t + 1) * Bn + tid) * Hn + c0], 0, 1);

    // epilogue: h_t = tanh(acc + xproj + bh); xproj lives in d_out[t]
    for (int nb = 0; nb < 2; ++nb) {
      int col = c0 + nb * 16 + lm;
      for (int r = 0; r < 8; ++r) {
        int row = wave * 16 + r + hi * 8;  // batch index
        size_t oidx = ((size_t)t * Bn + row) * Hn + col;
        float v = tanhf(acc[nb][r] + out[oidx] + bias[nb]);
        out[oidx] = v;
        hdst[(size_t)row * Hn + col] = f2bf(v);
      }
    }

    // grid barrier across the 16 persistent WGs (monotonic counter)
    __threadfence();
    __syncthreads();
    if (tid == 0) {
      atomicAdd(syncc, 1);
      const int target = (t + 1) * NWG;
      while (atomicAdd(syncc, 0) < target) __builtin_amdgcn_s_sleep(2);
    }
    __syncthreads();
    __threadfence();
  }
}

extern "C" void kernel_launch(void* const* d_in, const int* in_sizes, int n_in,
                              void* d_out, int out_size, void* d_ws,
                              size_t ws_size, hipStream_t stream) {
  (void)in_sizes; (void)n_in; (void)out_size; (void)ws_size;
  const float* x  = (const float*)d_in[0];
  const float* h0 = (const float*)d_in[1];
  const float* Wx = (const float*)d_in[2];
  const float* Wh = (const float*)d_in[3];
  const float* bh = (const float*)d_in[4];
  float* out = (float*)d_out;

  unsigned short* wx_bf = (unsigned short*)d_ws;       // 512*256
  unsigned short* wh_bf = wx_bf + Hn * In;             // 512*512
  unsigned short* hbuf0 = wh_bf + (size_t)Hn * Hn;     // 128*512
  unsigned short* hbuf1 = hbuf0 + (size_t)Bn * Hn;     // 128*512
  int* syncc = (int*)(hbuf1 + (size_t)Bn * Hn);

  (void)hipFuncSetAttribute((const void*)rnn_kernel,
                            hipFuncAttributeMaxDynamicSharedMemorySize,
                            P2SMEM);

  prep_kernel<<<(Hn * Hn + 255) / 256, 256, 0, stream>>>(h0, Wx, Wh, wx_bf,
                                                         wh_bf, hbuf0, syncc);
  xproj_kernel<<<(Lx * Bn) / 32, 256, 0, stream>>>(x, wx_bf, out);
  rnn_kernel<<<NWG, 256, P2SMEM, stream>>>(bh, wh_bf, hbuf0, hbuf1, out,
                                           syncc);
}
